// NWHead_66331474920046
// MI455X (gfx1250) — matrix operand order repaired and checked by reading, not compiled, not run on hardware
//
#include <hip/hip_runtime.h>
#include <math.h>

#define FEAT   1024
#define PROJ   256
#define BATCH  32
#define NSUP   8192
#define NROWS  (BATCH + NSUP)   // 8224 rows in P = [xq ; sxp]
#define NCLS   1000

typedef __attribute__((ext_vector_type(16))) __bf16 v16bf;
typedef __attribute__((ext_vector_type(8)))  float  v8f;
typedef __attribute__((ext_vector_type(4)))  float  v4f;

// ---------- helpers ----------

__device__ __forceinline__ __bf16 bf16_trunc(float f) {
    unsigned int u = __builtin_bit_cast(unsigned int, f);
    unsigned short hs = (unsigned short)(u >> 16);
    return __builtin_bit_cast(__bf16, hs);
}
__device__ __forceinline__ float bf16_to_f32(__bf16 h) {
    unsigned short hs = __builtin_bit_cast(unsigned short, h);
    unsigned int u = ((unsigned int)hs) << 16;
    return __builtin_bit_cast(float, u);
}

__device__ __forceinline__ v8f wmma_bf16(v16bf a, v16bf b, v8f c) {
    // (neg_a, A, neg_b, B, c_mod, C, reuse_a, reuse_b)
    return __builtin_amdgcn_wmma_f32_16x16x32_bf16(
        false, a, false, b, (short)0, c, false, false);
}

// split 16 f32 values into hi/lo bf16 fragments
__device__ __forceinline__ void split16(const float* av, v16bf* ah, v16bf* al) {
#pragma unroll
    for (int i = 0; i < 16; ++i) {
        __bf16 hi = bf16_trunc(av[i]);
        (*ah)[i] = hi;
        (*al)[i] = bf16_trunc(av[i] - bf16_to_f32(hi));
    }
}

// ---------- kernel 0: pack W (1024x256 f32) into WMMA B-fragment bf16 hi/lo planes ----------
// layout: [ktile(32)][ntile(16)][lane(32)][j(16)] ; lane = (n&15) + 16*((kin>>3)&1), j = (kin&7) + 8*(kin>>4)

__global__ void __launch_bounds__(256)
pack_w_kernel(const float* __restrict__ W, __bf16* __restrict__ Wh, __bf16* __restrict__ Wl) {
    int idx = blockIdx.x * 256 + threadIdx.x;   // k*256 + n
    int k = idx >> 8;
    int n = idx & 255;
    float w = W[idx];
    __bf16 hi = bf16_trunc(w);
    __bf16 lo = bf16_trunc(w - bf16_to_f32(hi));
    int ktile = k >> 5, kin = k & 31;
    int h = (kin >> 3) & 1, j = (kin & 7) + 8 * (kin >> 4);
    int ntile = n >> 4;
    int lane = (n & 15) + (h << 4);
    int out = (((ktile * 16) + ntile) * 32 + lane) * 16 + j;
    Wh[out] = hi;
    Wl[out] = lo;
}

// ---------- kernel 1: P = [x ; sx] @ W  via 3xBF16 WMMA ----------
// grid.x = 514 M-tiles, block = (32,4): wave nq handles N columns [64*nq, 64*nq+63]

__global__ void __launch_bounds__(128)
gemm_proj_kernel(const float* __restrict__ x, const float* __restrict__ sx,
                 const __bf16* __restrict__ Wh, const __bf16* __restrict__ Wl,
                 float* __restrict__ P) {
    const int lane = threadIdx.x;         // 0..31
    const int nq   = threadIdx.y;         // 0..3
    const int mt   = blockIdx.x;          // 0..513
    const int h    = lane >> 4;
    const int m    = lane & 15;
    const int row  = mt * 16 + m;
    const float* arow = (row < BATCH) ? (x + (size_t)row * FEAT)
                                      : (sx + (size_t)(row - BATCH) * FEAT);

    v8f acc[4];
#pragma unroll
    for (int t = 0; t < 4; ++t)
#pragma unroll
        for (int i = 0; i < 8; ++i) acc[t][i] = 0.0f;

    for (int ks = 0; ks < FEAT / 32; ++ks) {
        const int k0 = ks * 32;
        if (ks + 1 < FEAT / 32)
            __builtin_prefetch(arow + k0 + 32, 0, 3);   // global_prefetch_b8

        // A fragment (documented 16-bit A 16x32 layout):
        // lanes 0-15: K = {k0..k0+7, k0+16..k0+23}; lanes 16-31: +8
        const float* p0 = arow + k0 + 8 * h;
        v4f f0 = *(const v4f*)(p0);
        v4f f1 = *(const v4f*)(p0 + 4);
        v4f f2 = *(const v4f*)(p0 + 16);
        v4f f3 = *(const v4f*)(p0 + 20);
        float av[16];
#pragma unroll
        for (int i = 0; i < 4; ++i) {
            av[i] = f0[i]; av[4 + i] = f1[i]; av[8 + i] = f2[i]; av[12 + i] = f3[i];
        }
        v16bf ah, al;
        split16(av, &ah, &al);

#pragma unroll
        for (int t = 0; t < 4; ++t) {
            const int ntile = nq * 4 + t;
            const int fb = (((ks * 16) + ntile) * 32 + lane) * 16;
            v16bf bh = *(const v16bf*)(Wh + fb);
            v16bf bl = *(const v16bf*)(Wl + fb);
            acc[t] = wmma_bf16(al, bh, acc[t]);
            acc[t] = wmma_bf16(ah, bl, acc[t]);
            acc[t] = wmma_bf16(ah, bh, acc[t]);
        }
    }

    // C layout: VGPR v, lane L -> M = 8*(L>>4)+v, N = L&15
#pragma unroll
    for (int t = 0; t < 4; ++t) {
        const int col = nq * 64 + t * 16 + m;
#pragma unroll
        for (int v = 0; v < 8; ++v) {
            const int r = mt * 16 + h * 8 + v;
            P[(size_t)r * PROJ + col] = acc[t][v];
        }
    }
}

// ---------- kernel 2: squared row norms of P ----------
// block = (32,8): one wave per row

__global__ void __launch_bounds__(256)
rownorm_kernel(const float* __restrict__ P, float* __restrict__ sq) {
    const int lane = threadIdx.x;
    const int row  = blockIdx.x * 8 + threadIdx.y;
    const float* pr = P + (size_t)row * PROJ + lane * 8;
    v4f a = *(const v4f*)(pr);
    v4f b = *(const v4f*)(pr + 4);
    float s = 0.0f;
#pragma unroll
    for (int i = 0; i < 4; ++i) s += a[i] * a[i] + b[i] * b[i];
#pragma unroll
    for (int off = 16; off > 0; off >>= 1) s += __shfl_xor(s, off, 32);
    if (lane == 0) sq[row] = s;
}

// ---------- kernel 3: S[32 x 8192] = 2*xq.sxp^T - |xq|^2 - |sxp|^2 ----------
// grid.x = 128, block = (32,4): wave handles one 16-col N-tile, both 16-row M-tiles

__global__ void __launch_bounds__(128)
scores_kernel(const float* __restrict__ P, const float* __restrict__ sq,
              float* __restrict__ S) {
    const int lane = threadIdx.x;
    const int nt   = blockIdx.x * 4 + threadIdx.y;    // 0..511
    const int h    = lane >> 4;
    const int m    = lane & 15;

    v8f acc[2];
#pragma unroll
    for (int q = 0; q < 2; ++q)
#pragma unroll
        for (int i = 0; i < 8; ++i) acc[q][i] = 0.0f;

    const float* brow = P + (size_t)(BATCH + nt * 16 + m) * PROJ;

    for (int ks = 0; ks < PROJ / 32; ++ks) {
        const int k0 = ks * 32;

        // B fragment: lane half h holds K = k0+16h .. k0+16h+15 (contiguous in P row)
        const float* bp = brow + k0 + 16 * h;
        v4f b0 = *(const v4f*)(bp);
        v4f b1 = *(const v4f*)(bp + 4);
        v4f b2 = *(const v4f*)(bp + 8);
        v4f b3 = *(const v4f*)(bp + 12);
        float bv[16];
#pragma unroll
        for (int i = 0; i < 4; ++i) {
            bv[i] = b0[i]; bv[4 + i] = b1[i]; bv[8 + i] = b2[i]; bv[12 + i] = b3[i];
        }
        v16bf bh, bl;
        split16(bv, &bh, &bl);

#pragma unroll
        for (int q = 0; q < 2; ++q) {
            const float* ap = P + (size_t)(q * 16 + m) * PROJ + k0 + 16 * h;
            v4f f0 = *(const v4f*)(ap);
            v4f f1 = *(const v4f*)(ap + 4);
            v4f f2 = *(const v4f*)(ap + 8);
            v4f f3 = *(const v4f*)(ap + 12);
            float av[16];
#pragma unroll
            for (int i = 0; i < 4; ++i) {
                av[i] = f0[i]; av[4 + i] = f1[i]; av[8 + i] = f2[i]; av[12 + i] = f3[i];
            }
            v16bf ah, al;
            split16(av, &ah, &al);
            acc[q] = wmma_bf16(al, bh, acc[q]);
            acc[q] = wmma_bf16(ah, bl, acc[q]);
            acc[q] = wmma_bf16(ah, bh, acc[q]);
        }
    }

    const float ssq = sq[BATCH + nt * 16 + m];
#pragma unroll
    for (int q = 0; q < 2; ++q) {
#pragma unroll
        for (int v = 0; v < 8; ++v) {
            const int mrow = q * 16 + 8 * h + v;
            float val = 2.0f * acc[q][v] - sq[mrow] - ssq;
            S[(size_t)mrow * NSUP + nt * 16 + m] = val;
        }
    }
}

// ---------- kernel 4: per-row softmax + one-hot scatter (LDS bincount) + log ----------

__global__ void __launch_bounds__(256)
softmax_scatter_kernel(const float* __restrict__ S, const int* __restrict__ sy,
                       float* __restrict__ out) {
    __shared__ float red[256];
    __shared__ float bins[NCLS];
    const int b = blockIdx.x;
    const int tid = threadIdx.x;
    const float* srow = S + (size_t)b * NSUP;

    // max
    float mx = -3.402823466e38f;
    for (int j = tid; j < NSUP; j += 256) mx = fmaxf(mx, srow[j]);
    red[tid] = mx; __syncthreads();
    for (int s = 128; s > 0; s >>= 1) {
        if (tid < s) red[tid] = fmaxf(red[tid], red[tid + s]);
        __syncthreads();
    }
    mx = red[0]; __syncthreads();

    // sum of exp
    float sum = 0.0f;
    for (int j = tid; j < NSUP; j += 256) sum += __expf(srow[j] - mx);
    red[tid] = sum; __syncthreads();
    for (int s = 128; s > 0; s >>= 1) {
        if (tid < s) red[tid] += red[tid + s];
        __syncthreads();
    }
    const float inv = 1.0f / red[0];

    // scatter into class bins (probs @ one_hot == bincount of probs by label)
    for (int c = tid; c < NCLS; c += 256) bins[c] = 0.0f;
    __syncthreads();
    for (int j = tid; j < NSUP; j += 256) {
        float p = __expf(srow[j] - mx) * inv;
        atomicAdd(&bins[sy[j]], p);   // ds_add_f32
    }
    __syncthreads();
    for (int c = tid; c < NCLS; c += 256)
        out[(size_t)b * NCLS + c] = logf(bins[c] + 1e-12f);
}

// ---------- launcher ----------

extern "C" void kernel_launch(void* const* d_in, const int* in_sizes, int n_in,
                              void* d_out, int out_size, void* d_ws, size_t ws_size,
                              hipStream_t stream) {
    (void)in_sizes; (void)n_in; (void)out_size; (void)ws_size;
    const float* x  = (const float*)d_in[0];   // (32, 1024)
    const float* sx = (const float*)d_in[1];   // (8192, 1024)
    const float* W  = (const float*)d_in[2];   // (1024, 256)
    const int*   sy = (const int*)d_in[3];     // (8192,)
    float* out = (float*)d_out;                // (32, 1000)

    char* ws = (char*)d_ws;
    // offsets (256B aligned)
    constexpr size_t OFF_P  = 0;                                   // 8224*256*4  = 8421376
    constexpr size_t OFF_WH = OFF_P  + (size_t)NROWS * PROJ * 4;   // +8421376
    constexpr size_t OFF_WL = OFF_WH + (size_t)FEAT * PROJ * 2;    // +524288
    constexpr size_t OFF_SQ = OFF_WL + (size_t)FEAT * PROJ * 2;    // +524288
    constexpr size_t OFF_S  = OFF_SQ + 33280;                      // 8224*4 rounded up
    float*  P  = (float*)(ws + OFF_P);
    __bf16* Wh = (__bf16*)(ws + OFF_WH);
    __bf16* Wl = (__bf16*)(ws + OFF_WL);
    float*  sq = (float*)(ws + OFF_SQ);
    float*  S  = (float*)(ws + OFF_S);

    pack_w_kernel<<<(FEAT * PROJ) / 256, 256, 0, stream>>>(W, Wh, Wl);
    gemm_proj_kernel<<<NROWS / 16, dim3(32, 4), 0, stream>>>(x, sx, Wh, Wl, P);
    rownorm_kernel<<<NROWS / 8, dim3(32, 8), 0, stream>>>(P, sq);
    scores_kernel<<<NSUP / 16 / 4, dim3(32, 4), 0, stream>>>(P, sq, S);
    softmax_scatter_kernel<<<BATCH, 256, 0, stream>>>(S, sy, out);
}
